// Block_68358699483301
// MI455X (gfx1250) — matrix operand order repaired and checked by reading, not compiled, not run on hardware
//
#include <hip/hip_runtime.h>
#include <hip/hip_bf16.h>
#include <math.h>

#define BB     2
#define NN     1536
#define MM     (BB * NN)
#define DEG    32
#define AD     128
#define VD     32
#define RK     32
#define NH     16
#define DHD    8

typedef __attribute__((ext_vector_type(16))) _Float16 v16h;
typedef __attribute__((ext_vector_type(8)))  float    v8f;
typedef __attribute__((ext_vector_type(8)))  _Float16 h8;
typedef __attribute__((ext_vector_type(4)))  unsigned int u32x4;
typedef __attribute__((ext_vector_type(4)))  int      i32x4;
typedef __attribute__((ext_vector_type(8)))  int      i32x8;

#ifndef __has_builtin
#define __has_builtin(x) 0
#endif
#if __has_builtin(__builtin_amdgcn_tensor_load_to_lds) && \
    __has_builtin(__builtin_amdgcn_s_wait_tensorcnt)
#define USE_TDM 1
#else
#define USE_TDM 0
#endif

// ---------------------------------------------------------------------------
// WMMA helpers (CDNA5 wave32, 16x16x32 f16 -> f32)
// ---------------------------------------------------------------------------
__device__ __forceinline__ v8f wmma16(v16h a, v16h b, v8f c) {
  // 8 args: (neg_a, A, neg_b, B, c_mod, C, reuse_a, reuse_b)
  return __builtin_amdgcn_wmma_f32_16x16x32_f16(false, a, false, b, (short)0, c,
                                                false, false);
}

// A-matrix 16x32 f16 layout: lane holds row (lane&15); K split:
// lanes 0-15: elems 0-7 -> K 0-7, elems 8-15 -> K 16-23
// lanes 16-31: elems 0-7 -> K 8-15, elems 8-15 -> K 24-31
__device__ __forceinline__ v16h frag_a_f32(const float* A, int lda, int r0,
                                           int k0, int lane) {
  int row = r0 + (lane & 15);
  int kb  = k0 + 8 * (lane >> 4);
  const float* p = A + (size_t)row * lda + kb;
  v16h a;
#pragma unroll
  for (int e = 0; e < 8; ++e) a[e] = (_Float16)p[e];
#pragma unroll
  for (int e = 0; e < 8; ++e) a[8 + e] = (_Float16)p[16 + e];
  return a;
}

// B-matrix 32x16 f16 layout: lane holds col (lane&15);
// lanes 0-15: elems 0-7 -> K 0-7, elems 8-15 -> K 16-23; lanes 16-31: K 8-15, 24-31
__device__ __forceinline__ v16h frag_b_f32(const float* W, int ldb, int k0,
                                           int n0, int lane) {
  int n  = n0 + (lane & 15);
  int kb = k0 + 8 * (lane >> 4);
  v16h b;
#pragma unroll
  for (int e = 0; e < 16; ++e) b[e] = (_Float16)W[(size_t)(kb + e + (e & 8)) * ldb + n];
  return b;
}

__device__ __forceinline__ float leaky(float x) {
  return x >= 0.f ? x : 0.2f * x;
}

// ---------------------------------------------------------------------------
// Kernel A: graph embed + local residual + group norms + probes (per node)
// block = 128 threads (4 waves), grid = MM
// ---------------------------------------------------------------------------
__global__ void embed_local_kernel(
    const float* __restrict__ x_a, const float* __restrict__ x_v,
    const float* __restrict__ pos, const float* __restrict__ box,
    const int* __restrict__ nbrs,
    const float* __restrict__ emb_a_w, const float* __restrict__ emb_a_b,
    const float* __restrict__ emb_v_w,
    const float* __restrict__ la1_w, const float* __restrict__ la1_b,
    const float* __restrict__ la2_w, const float* __restrict__ la2_b,
    const float* __restrict__ la3_w, const float* __restrict__ la3_b,
    const float* __restrict__ lin_v_w,
    const float* __restrict__ avp_a_w, const float* __restrict__ avp_a_b,
    const float* __restrict__ avp_v_w,
    const float* __restrict__ avp_ao_w, const float* __restrict__ avp_ao_b,
    const float* __restrict__ avp_vo_w,
    const float* __restrict__ gna_g, const float* __restrict__ gna_b,
    const float* __restrict__ gnv_g,
    const float* __restrict__ pq_w, const float* __restrict__ pk_w,
    float* __restrict__ z_a, float* __restrict__ z_v,
    float* __restrict__ prq, float* __restrict__ prk) {
  __shared__ float ya[AD], yv[VD * 3], hA[AD], hB[AD];
  __shared__ float ea[8], ev[24];
  __shared__ float ar[RK], vr[RK * 3], sr[RK];
  __shared__ float us[AD], gm[8], gv[8];
  __shared__ float wv[VD * 3], gms[8], zvl[VD * 3];

  const int i = blockIdx.x;            // global node index (b*NN + n)
  const int b = i / NN, n = i % NN;
  const int t = threadIdx.x;           // 0..127
  const float ab = box[0] * 0.15915494309189535f;  // box / 2pi

  if (t < 8) ea[t] = 0.f;
  if (t < 24) ev[t] = 0.f;
  __syncthreads();

  // --- neighbor gather + radial encode + reduce ---
  if (t < DEG) {
    int nb = nbrs[n * DEG + t];
    float e[3], r2 = 0.f;
#pragma unroll
    for (int c = 0; c < 3; ++c) {
      float d = pos[(size_t)(b * NN + nb) * 3 + c] - pos[(size_t)i * 3 + c];
      e[c] = ab * __sinf(d / ab);
      r2 += e[c] * e[c];
    }
    float d = sqrtf(r2 + 1e-6f);
    float cut = 0.5f * (__cosf(3.14159265358979f * fminf(d * (1.f / 3.f), 1.f)) + 1.f);
#pragma unroll
    for (int k = 0; k < 8; ++k) {
      float ck = (3.0f / 7.0f) * (float)k;
      float zz = (d - ck) * (1.0f / 0.375f);
      float x = __expf(-0.5f * zz * zz) * cut;
      atomicAdd(&ea[k], x);
#pragma unroll
      for (int c = 0; c < 3; ++c) atomicAdd(&ev[k * 3 + c], x * e[c]);
    }
  }
  __syncthreads();

  // --- y_a, y_v ---
  {
    float s = x_a[(size_t)i * AD + t] + emb_a_b[t];
#pragma unroll
    for (int k = 0; k < 8; ++k) s += ea[k] * emb_a_w[k * AD + t];
    ya[t] = s;
  }
  if (t < VD * 3) {
    int o = t / 3, c = t % 3;
    float s = x_v[(size_t)i * VD * 3 + t];
#pragma unroll
    for (int k = 0; k < 8; ++k) s += ev[k * 3 + c] * emb_v_w[o * 8 + k];
    yv[t] = s;
  }
  __syncthreads();

  // --- scalar MLP (3x 128x128 matvec per node) ---
  {
    float s = la1_b[t];
    for (int k = 0; k < AD; ++k) s += ya[k] * la1_w[k * AD + t];
    hA[t] = leaky(s);
  }
  __syncthreads();
  {
    float s = la2_b[t];
    for (int k = 0; k < AD; ++k) s += hA[k] * la2_w[k * AD + t];
    hB[t] = leaky(s);
  }
  __syncthreads();
  float ya2 = la3_b[t];
  for (int k = 0; k < AD; ++k) ya2 += hB[k] * la3_w[k * AD + t];

  // --- low-rank scalar<->vector coupling ---
  if (t < RK) {
    float s = avp_a_b[t];
    for (int k = 0; k < AD; ++k) s += ya[k] * avp_a_w[k * RK + t];
    ar[t] = s;
  }
  if (t < RK * 3) {
    int r = t / 3, c = t % 3;
    float s = 0.f;
#pragma unroll
    for (int k = 0; k < VD; ++k) s += yv[k * 3 + c] * avp_v_w[r * VD + k];
    vr[t] = s;
  }
  __syncthreads();
  if (t < RK) {
    float q = vr[t * 3] * vr[t * 3] + vr[t * 3 + 1] * vr[t * 3 + 1] +
              vr[t * 3 + 2] * vr[t * 3 + 2];
    sr[t] = ar[t] * q;
  }
  __syncthreads();
  float pa = avp_ao_b[t];
#pragma unroll
  for (int r = 0; r < RK; ++r) pa += sr[r] * avp_ao_w[r * AD + t];

  // --- scalar group norm (8 groups x 16) ---
  float u = ya2 + pa;
  us[t] = u;
  __syncthreads();
  if (t < 8) {
    float mu = 0.f;
    for (int j = 0; j < 16; ++j) mu += us[t * 16 + j];
    mu *= (1.f / 16.f);
    float vv2 = 0.f;
    for (int j = 0; j < 16; ++j) {
      float d = us[t * 16 + j] - mu;
      vv2 += d * d;
    }
    gm[t] = mu;
    gv[t] = vv2 * (1.f / 16.f);
  }
  __syncthreads();
  {
    int g = t >> 4;
    float zz = (u - gm[g]) * rsqrtf(gv[g] + 1e-5f);
    z_a[(size_t)i * AD + t] = ya[t] + zz * gna_g[t] + gna_b[t];
  }

  // --- vector branch: yv2 + p_v, vector group norm ---
  if (t < VD * 3) {
    int o = t / 3, c = t % 3;
    float s = 0.f;
#pragma unroll
    for (int k = 0; k < VD; ++k) s += yv[k * 3 + c] * lin_v_w[o * VD + k];
    float pv = 0.f;
#pragma unroll
    for (int r = 0; r < RK; ++r) pv += ar[r] * vr[r * 3 + c] * avp_vo_w[o * RK + r];
    wv[t] = s + pv;
  }
  __syncthreads();
  if (t < 8) {
    float s = 0.f;
    for (int j = 0; j < 12; ++j) {
      float x = wv[t * 12 + j];
      s += x * x;
    }
    gms[t] = s * 0.25f;  // mean sq-norm over 4 channels
  }
  __syncthreads();
  if (t < VD * 3) {
    int o = t / 3, g = o >> 2;
    float z = yv[t] + wv[t] * rsqrtf(gms[g] + 1e-5f) * gnv_g[o];
    z_v[(size_t)i * VD * 3 + t] = z;
    zvl[t] = z;
  }
  __syncthreads();

  // --- probe points ---
  if (t < 12) {
    int which = t / 6, p = (t % 6) / 3, c = t % 3;
    const float* w = which ? pk_w : pq_w;
    float s = pos[(size_t)i * 3 + c];
#pragma unroll
    for (int k = 0; k < VD; ++k) s += zvl[k * 3 + c] * w[p * VD + k];
    (which ? prk : prq)[(size_t)i * 6 + p * 3 + c] = s;
  }
}

// ---------------------------------------------------------------------------
// Kernel B: q/k/va projections via WMMA. q,k -> f16 row-major [row*128];
// va -> packed value records vpk[row*256 + head*16 + dim] (dims 0-7).
// grid = (MM/16, 8, 3), block = 32 (1 wave)
// ---------------------------------------------------------------------------
__global__ void qkva_kernel(const float* __restrict__ za,
                            const float* __restrict__ wq,
                            const float* __restrict__ wk,
                            const float* __restrict__ wva,
                            _Float16* __restrict__ qh, _Float16* __restrict__ kh,
                            _Float16* __restrict__ vpk) {
  int lane = threadIdx.x;
  int r0 = blockIdx.x * 16;
  int n0 = blockIdx.y * 16;
  int mat = blockIdx.z;
  const float* W = (mat == 0) ? wq : (mat == 1 ? wk : wva);
  v8f c = {};
#pragma unroll
  for (int k0 = 0; k0 < AD; k0 += 32) {
    v16h a = frag_a_f32(za, AD, r0, k0, lane);
    v16h b = frag_b_f32(W, AD, k0, n0, lane);
    c = wmma16(a, b, c);
  }
  int n = n0 + (lane & 15);
  int rb = r0 + 8 * (lane >> 4);
  if (mat < 2) {
    _Float16* O = (mat == 0) ? qh : kh;
#pragma unroll
    for (int v = 0; v < 8; ++v) O[(size_t)(rb + v) * AD + n] = (_Float16)c[v];
  } else {
#pragma unroll
    for (int v = 0; v < 8; ++v)
      vpk[(size_t)(rb + v) * 256 + (n >> 3) * 16 + (n & 7)] = (_Float16)c[v];
  }
}

// ---------------------------------------------------------------------------
// Kernel Bv: vv projection per component (M x 32 @ 32 x 32) via WMMA.
// Writes packed value records: vpk[row*256 + head*16 + 8 + cw*3 + comp].
// grid = (MM/16, 3 comps, 2 col tiles), block = 32
// ---------------------------------------------------------------------------
__global__ void vvproj_kernel(const float* __restrict__ zv,
                              const float* __restrict__ wvv,
                              _Float16* __restrict__ vpk) {
  int lane = threadIdx.x;
  int r0 = blockIdx.x * 16;
  int comp = blockIdx.y;
  int n0 = blockIdx.z * 16;
  int row = r0 + (lane & 15);
  int kbA = 8 * (lane >> 4);
  v16h a, b;
#pragma unroll
  for (int e = 0; e < 8; ++e) {
    a[e] = (_Float16)zv[(size_t)row * (VD * 3) + (kbA + e) * 3 + comp];
    a[8 + e] = (_Float16)zv[(size_t)row * (VD * 3) + (kbA + 16 + e) * 3 + comp];
  }
  int n = n0 + (lane & 15);
  int kbB = 8 * (lane >> 4);
#pragma unroll
  for (int e = 0; e < 16; ++e) b[e] = (_Float16)wvv[n * VD + kbB + e + (e & 8)];  // W^T[k][n]
  v8f c = {};
  c = wmma16(a, b, c);
  int rb = r0 + 8 * (lane >> 4);
#pragma unroll
  for (int v = 0; v < 8; ++v) {
    int o = n, q = rb + v;
    vpk[(size_t)q * 256 + (o >> 1) * 16 + 8 + (o & 1) * 3 + comp] = (_Float16)c[v];
  }
}

// Zero the 2 pad columns of each packed value record (ws is poisoned 0xAA).
__global__ void vpad_kernel(_Float16* __restrict__ vpk) {
  int i = blockIdx.x;
  int t = threadIdx.x;  // 0..31: (head, slot)
  vpk[(size_t)i * 256 + (t >> 1) * 16 + 14 + (t & 1)] = (_Float16)0.f;
}

// ---------------------------------------------------------------------------
// Kernel C: fused proximity flash attention. One wave per (b, head, 16-query
// tile). 32 keys per iteration: 2 QK^T WMMAs + online softmax + 1 P*(Va|Vv)
// WMMA. The 32x16-half packed value tile is DMA'd to LDS by the Tensor Data
// Mover (tensor_load_to_lds + s_wait_tensorcnt) when the builtin exists.
// grid = (NN/16, NH, BB), block = 32
// ---------------------------------------------------------------------------
__global__ void attn_kernel(const _Float16* __restrict__ qh,
                            const _Float16* __restrict__ kh,
                            const _Float16* __restrict__ vpk,
                            const float* __restrict__ prq,
                            const float* __restrict__ prk,
                            const float* __restrict__ box,
                            float* __restrict__ oa, float* __restrict__ ov) {
  __shared__ float pqs[16][3];
  __shared__ __align__(16) _Float16 pl[16 * 32];
#if USE_TDM
  __shared__ __align__(16) _Float16 vtile[32 * 16];
#endif

  const int lane = threadIdx.x;
  const int q0 = blockIdx.x * 16;
  const int h = blockIdx.y;
  const int b = blockIdx.z;
  const int base = b * NN;
  const float ab = box[0] * 0.15915494309189535f;
  const int p = h >> 3;                       // repeat(kern, NH/NPROBE)
  const float r0v = (p == 0) ? 1.0f : 2.0f;   // R0_LIST
  const float negi = -1.0f / (2.0f * r0v * r0v);
  const float iss = 0.35355339059327373f;     // 1/sqrt(DH)
  const int col = lane & 15;
  const int half = lane >> 4;

  if (lane < 16) {
#pragma unroll
    for (int c = 0; c < 3; ++c)
      pqs[lane][c] = prq[(size_t)(base + q0 + lane) * 6 + p * 3 + c];
  }
  __syncthreads();

  // Q fragment: DH=8 real values zero-padded into the K=32 contraction.
  v16h qa = {};
  if (lane < 16) {
    h8 qv = *(const h8*)&qh[(size_t)(base + q0 + lane) * AD + h * DHD];
#pragma unroll
    for (int e = 0; e < 8; ++e) qa[e] = qv[e];
  }

  v8f acc = {};
  float mi[8], li[8];
#pragma unroll
  for (int v = 0; v < 8; ++v) { mi[v] = -1e30f; li[v] = 0.f; }

  for (int kb = 0; kb < NN; kb += 32) {
#if USE_TDM
    {
      // Tensor DMA descriptor (D#) per CDNA5 ISA 8.3/8.4:
      // 2D tile: 16 elems (32B) x 32 rows from vpk[(base+kb)..][h*16..]
      unsigned int lds_off = (unsigned int)(uintptr_t)(void*)vtile;
      unsigned long long ga =
          (unsigned long long)(uintptr_t)vpk +
          (((unsigned long long)(base + kb)) * 256ull +
           (unsigned long long)h * 16ull) * 2ull;
      u32x4 g0;
      g0[0] = 1u;                                   // count=1 (user mode)
      g0[1] = lds_off;                              // lds_addr
      g0[2] = (unsigned int)ga;                     // global_addr[31:0]
      g0[3] = (unsigned int)((ga >> 32) & 0x01FFFFFFull) | 0x80000000u;
                                                    // addr[56:32] | type=2
      i32x8 g1;
      g1[0] = 0x00010000;                           // data_size=1 (2 bytes)
      g1[1] = (int)((256u & 0xFFFFu) << 16);        // tensor_dim0 lo16
      g1[2] = (int)((256u >> 16) & 0xFFFFu) |
              (int)(((unsigned)MM & 0xFFFFu) << 16); // dim0 hi | tensor_dim1 lo
      g1[3] = (int)(((unsigned)MM >> 16) & 0xFFFFu) |
              (16 << 16);                           // dim1 hi | tile_dim0=16
      g1[4] = 32;                                   // tile_dim1 = 32 rows
      g1[5] = 256;                                  // tensor_dim0_stride lo32
      g1[6] = 0;
      g1[7] = 0;
      i32x4 z4 = {0, 0, 0, 0};
#if defined(__clang_major__) && __clang_major__ >= 23
      i32x8 z8 = {0, 0, 0, 0, 0, 0, 0, 0};
      __builtin_amdgcn_tensor_load_to_lds(g0, g1, z4, z4, z8, 0);
#else
      __builtin_amdgcn_tensor_load_to_lds(g0, g1, z4, z4, 0);
#endif
    }
#endif

    // K^T fragments for two 16-key tiles (B layout: lane = key col)
    v16h b0 = {}, b1 = {};
    if (lane < 16) {
      h8 k0 = *(const h8*)&kh[(size_t)(base + kb + col) * AD + h * DHD];
      h8 k1 = *(const h8*)&kh[(size_t)(base + kb + 16 + col) * AD + h * DHD];
#pragma unroll
      for (int e = 0; e < 8; ++e) { b0[e] = k0[e]; b1[e] = k1[e]; }
    }
    v8f zc = {};
    v8f s0 = wmma16(qa, b0, zc);
    v8f s1 = wmma16(qa, b1, zc);

    // Gaussian proximity log-kernel on probe points (periodic sin wrap)
    int key0 = kb + col, key1 = kb + 16 + col;
    float pk0[3], pk1[3];
#pragma unroll
    for (int c = 0; c < 3; ++c) {
      pk0[c] = prk[(size_t)(base + key0) * 6 + p * 3 + c];
      pk1[c] = prk[(size_t)(base + key1) * 6 + p * 3 + c];
    }
    float mx[8];
#pragma unroll
    for (int v = 0; v < 8; ++v) {
      int m = v + 8 * half;
      float r2a = 0.f, r2b = 0.f;
#pragma unroll
      for (int c = 0; c < 3; ++c) {
        float qc = pqs[m][c];
        float d0 = ab * __sinf((qc - pk0[c]) / ab);
        float d1 = ab * __sinf((qc - pk1[c]) / ab);
        r2a += d0 * d0;
        r2b += d1 * d1;
      }
      s0[v] = s0[v] * iss + r2a * negi;
      s1[v] = s1[v] * iss + r2b * negi;
      mx[v] = fmaxf(s0[v], s1[v]);
    }
    // row max: butterfly over 16 key-lanes (stays within each lane half)
#pragma unroll
    for (int v = 0; v < 8; ++v) {
      mx[v] = fmaxf(mx[v], __shfl_xor(mx[v], 1, 32));
      mx[v] = fmaxf(mx[v], __shfl_xor(mx[v], 2, 32));
      mx[v] = fmaxf(mx[v], __shfl_xor(mx[v], 4, 32));
      mx[v] = fmaxf(mx[v], __shfl_xor(mx[v], 8, 32));
    }
    float rs[8];
#pragma unroll
    for (int v = 0; v < 8; ++v) {
      float mnew = fmaxf(mi[v], mx[v]);
      float sc = __expf(mi[v] - mnew);
      s0[v] = __expf(s0[v] - mnew);
      s1[v] = __expf(s1[v] - mnew);
      rs[v] = s0[v] + s1[v];
      li[v] *= sc;
      acc[v] *= sc;
      mi[v] = mnew;
    }
#pragma unroll
    for (int v = 0; v < 8; ++v) {
      rs[v] += __shfl_xor(rs[v], 1, 32);
      rs[v] += __shfl_xor(rs[v], 2, 32);
      rs[v] += __shfl_xor(rs[v], 4, 32);
      rs[v] += __shfl_xor(rs[v], 8, 32);
      li[v] += rs[v];
    }

    // C-layout -> A-layout transpose of P through LDS
#pragma unroll
    for (int v = 0; v < 8; ++v) {
      int m = v + 8 * half;
      pl[m * 32 + col] = (_Float16)s0[v];
      pl[m * 32 + 16 + col] = (_Float16)s1[v];
    }
#if USE_TDM
    __builtin_amdgcn_s_wait_tensorcnt(0);
#endif
    __syncthreads();
    v16h pa;
    {
      int r = col, kk = 8 * half;
      h8 p0 = *(const h8*)&pl[r * 32 + kk];
      h8 p1 = *(const h8*)&pl[r * 32 + 16 + kk];
#pragma unroll
      for (int e = 0; e < 8; ++e) { pa[e] = p0[e]; pa[8 + e] = p1[e]; }
    }
    // packed value B tile: cols 0-7 = va head dims, 8-13 = vv, 14-15 = pad
    v16h bv;
#pragma unroll
    for (int e = 0; e < 16; ++e) {
      int key = e + (e & 8) + 8 * half;
#if USE_TDM
      bv[e] = vtile[key * 16 + col];
#else
      bv[e] = vpk[(size_t)(base + kb + key) * 256 + h * 16 + col];
#endif
    }
    acc = wmma16(pa, bv, acc);
    __syncthreads();

    // prefetch next key tile (emits global_prefetch_b8 on gfx1250)
    if (kb + 32 < NN && lane < 16)
      __builtin_prefetch(&kh[(size_t)(base + kb + 32 + col) * AD + h * DHD], 0, 1);
  }

  // finalize: divide by softmax denominator, scatter out_a / out_v
#pragma unroll
  for (int v = 0; v < 8; ++v) {
    float o = acc[v] / li[v];
    int q = q0 + v + 8 * half;
    if (col < 8)
      oa[(size_t)(base + q) * AD + h * DHD + col] = o;
    else if (col < 14)
      ov[(size_t)(base + q) * 96 + h * 6 + (col - 8)] = o;
  }
}

// ---------------------------------------------------------------------------
// Kernel D1: z_a_out = z_a + out_a @ wo_a + b  (WMMA), writes out[:, 0:128]
// grid = (MM/16, 8), block = 32
// ---------------------------------------------------------------------------
__global__ void outa_kernel(const float* __restrict__ oa,
                            const float* __restrict__ wo_a,
                            const float* __restrict__ wo_b,
                            const float* __restrict__ za,
                            float* __restrict__ out) {
  int lane = threadIdx.x;
  int r0 = blockIdx.x * 16;
  int n0 = blockIdx.y * 16;
  v8f c = {};
#pragma unroll
  for (int k0 = 0; k0 < AD; k0 += 32) {
    v16h a = frag_a_f32(oa, AD, r0, k0, lane);
    v16h b = frag_b_f32(wo_a, AD, k0, n0, lane);
    c = wmma16(a, b, c);
  }
  int n = n0 + (lane & 15);
  int rb = r0 + 8 * (lane >> 4);
  float bias = wo_b[n];
#pragma unroll
  for (int v = 0; v < 8; ++v) {
    int r = rb + v;
    out[(size_t)r * 224 + n] = za[(size_t)r * AD + n] + c[v] + bias;
  }
}

// ---------------------------------------------------------------------------
// Kernel D2: z_v_out = z_v + vec_linear(out_v, wo_v), writes out[:, 128:224]
// grid = MM, block = 96
// ---------------------------------------------------------------------------
__global__ void outv_kernel(const float* __restrict__ ovv,
                            const float* __restrict__ wo_v,
                            const float* __restrict__ zv,
                            float* __restrict__ out) {
  int i = blockIdx.x;
  int t = threadIdx.x;  // 0..95: (o, c)
  int o = t / 3, c = t % 3;
  float s = 0.f;
#pragma unroll
  for (int k = 0; k < VD; ++k)
    s += ovv[(size_t)i * 96 + k * 3 + c] * wo_v[o * VD + k];
  out[(size_t)i * 224 + 128 + t] = zv[(size_t)i * 96 + t] + s;
}

// ---------------------------------------------------------------------------
extern "C" void kernel_launch(void* const* d_in, const int* in_sizes, int n_in,
                              void* d_out, int out_size, void* d_ws,
                              size_t ws_size, hipStream_t stream) {
  (void)in_sizes; (void)n_in; (void)out_size; (void)ws_size;
  const float* x_a     = (const float*)d_in[0];
  const float* x_v     = (const float*)d_in[1];
  const float* pos0    = (const float*)d_in[2];
  const float* box     = (const float*)d_in[3];
  const int*   nbrs    = (const int*)d_in[4];
  const float* emb_a_w = (const float*)d_in[5];
  const float* emb_a_b = (const float*)d_in[6];
  const float* emb_v_w = (const float*)d_in[7];
  const float* la1_w   = (const float*)d_in[8];
  const float* la1_b   = (const float*)d_in[9];
  const float* la2_w   = (const float*)d_in[10];
  const float* la2_b   = (const float*)d_in[11];
  const float* la3_w   = (const float*)d_in[12];
  const float* la3_b   = (const float*)d_in[13];
  const float* lin_v_w = (const float*)d_in[14];
  const float* avp_a_w = (const float*)d_in[15];
  const float* avp_a_b = (const float*)d_in[16];
  const float* avp_v_w = (const float*)d_in[17];
  const float* avp_ao_w= (const float*)d_in[18];
  const float* avp_ao_b= (const float*)d_in[19];
  const float* avp_vo_w= (const float*)d_in[20];
  const float* gna_g   = (const float*)d_in[21];
  const float* gna_b   = (const float*)d_in[22];
  const float* gnv_g   = (const float*)d_in[23];
  const float* pq_w    = (const float*)d_in[24];
  const float* pk_w    = (const float*)d_in[25];
  const float* wq      = (const float*)d_in[26];
  const float* wk      = (const float*)d_in[27];
  const float* wva     = (const float*)d_in[28];
  const float* wo_a    = (const float*)d_in[29];
  const float* wo_a_b  = (const float*)d_in[30];
  const float* wvv     = (const float*)d_in[31];
  const float* wo_v    = (const float*)d_in[32];
  float* out = (float*)d_out;

  char* ws = (char*)d_ws;
  size_t off = 0;
  auto take = [&](size_t bytes) -> void* {
    void* p = ws + off;
    off += bytes;
    off = (off + 255) & ~(size_t)255;
    return p;
  };
  float* z_a     = (float*)take((size_t)MM * AD * 4);
  float* z_v     = (float*)take((size_t)MM * VD * 3 * 4);
  float* prq     = (float*)take((size_t)MM * 6 * 4);
  float* prk     = (float*)take((size_t)MM * 6 * 4);
  _Float16* qh   = (_Float16*)take((size_t)MM * AD * 2);
  _Float16* kh   = (_Float16*)take((size_t)MM * AD * 2);
  _Float16* vpk  = (_Float16*)take((size_t)MM * NH * 16 * 2);
  float* oa      = (float*)take((size_t)MM * AD * 4);
  float* ov      = (float*)take((size_t)MM * 96 * 4);

  embed_local_kernel<<<MM, 128, 0, stream>>>(
      x_a, x_v, pos0, box, nbrs, emb_a_w, emb_a_b, emb_v_w, la1_w, la1_b,
      la2_w, la2_b, la3_w, la3_b, lin_v_w, avp_a_w, avp_a_b, avp_v_w,
      avp_ao_w, avp_ao_b, avp_vo_w, gna_g, gna_b, gnv_g, pq_w, pk_w,
      z_a, z_v, prq, prk);

  qkva_kernel<<<dim3(MM / 16, AD / 16, 3), 32, 0, stream>>>(z_a, wq, wk, wva,
                                                            qh, kh, vpk);
  vvproj_kernel<<<dim3(MM / 16, 3, 2), 32, 0, stream>>>(z_v, wvv, vpk);
  vpad_kernel<<<MM, 32, 0, stream>>>(vpk);

  attn_kernel<<<dim3(NN / 16, NH, BB), 32, 0, stream>>>(qh, kh, vpk, prq, prk,
                                                        box, oa, ov);

  outa_kernel<<<dim3(MM / 16, AD / 16), 32, 0, stream>>>(oa, wo_a, wo_a_b, z_a,
                                                         out);
  outv_kernel<<<MM, 96, 0, stream>>>(ov, wo_v, z_v, out);
}
